// VectorNet_14362370638063
// MI455X (gfx1250) — matrix operand order repaired and checked by reading, not compiled, not run on hardware
//
#include <hip/hip_runtime.h>
#include <math.h>

// ---------------------------------------------------------------------------
// CDNA5 (gfx1250) VectorNet forward.  wave32 WMMA f16->f32 for all dense
// layers (operands staged as padded f16 so the hot loop is pure b128 loads +
// wmma); float atomics for GATv2 segment softmax / scatter.
// ---------------------------------------------------------------------------

typedef __attribute__((ext_vector_type(16))) _Float16 v16h;
typedef __attribute__((ext_vector_type(8)))  _Float16 v8h;
typedef __attribute__((ext_vector_type(8)))  float    v8f;

#define NEG_SLOPE 0.2f

static inline int padK(int K) { return (K + 31) & ~31; }

// ------------------- f16 staging pre-passes (tiny, once per layer) ----------
// XH[r, k] = (f16) X[r, k]  (k < K), 0 otherwise.   XH stride = Kp.
__global__ void cvt_rows_f16_kernel(const float* __restrict__ src,
                                    _Float16* __restrict__ dst,
                                    long long total, int K, int Kp)
{
    const long long i = blockIdx.x * (long long)blockDim.x + threadIdx.x;
    if (i >= total) return;
    const int       k = (int)(i % Kp);
    const long long r = i / Kp;
    dst[i] = (_Float16)((k < K) ? src[r * K + k] : 0.0f);
}

// WT[n, k] = (f16) W[k, n]  (transposed, zero-padded to Kp).
__global__ void cvt_wt_f16_kernel(const float* __restrict__ W,
                                  _Float16* __restrict__ WT,
                                  long long total, int K, int F, int Kp)
{
    const long long i = blockIdx.x * (long long)blockDim.x + threadIdx.x;
    if (i >= total) return;
    const int k = (int)(i % Kp);
    const int n = (int)(i / Kp);
    WT[i] = (_Float16)((k < K) ? W[(size_t)k * F + n] : 0.0f);
}

// ------------------------- WMMA tiled linear --------------------------------
// Y[Nr,F] = act(XH[Nr,Kp](f16) @ WT[F,Kp](f16)^T + bias).
// One wave computes a 64x16 output tile (4 stacked 16x16 WMMA tiles sharing
// each B fragment).  Per CDNA5 ISA 7.12.2 (wave32) the fragment layouts
// collapse, for contiguous K, to:
//  A: lane = 16*khalf + m ; afrag = XH[row, k0+8*khalf .. +7] ++ XH[row, k0+16+8*khalf .. +7]
//  B: lane = 16*khalf + n ; bfrag = WT[col, k0+8*khalf .. +7] ++ WT[col, k0+16+8*khalf .. +7]
//  D: lane = 16*mhalf + n ; VGPR r -> M = r + 8*mhalf
// Rows/cols are CLAMPED on load (garbage only reaches unstored D rows/cols);
// zero-padding of XH/WT removes any K-tail handling.
__global__ void linear_wmma_kernel(const _Float16* __restrict__ XH,
                                   const _Float16* __restrict__ WT,
                                   const float* __restrict__ bias,
                                   float* __restrict__ Y,
                                   int Nr, int Kp, int F, int relu)
{
    const int lane   = threadIdx.x & 31;
    const int wave   = threadIdx.x >> 5;
    const int i0     = blockIdx.x * 64;
    const int j0     = (blockIdx.y * 4 + wave) * 16;
    if (j0 >= F) return;

    const int half   = lane >> 4;      // K-half carried by this lane
    const int lane16 = lane & 15;

    const int bcol = min(j0 + lane16, F - 1);
    const int row0 = min(i0 +  0 + lane16, Nr - 1);
    const int row1 = min(i0 + 16 + lane16, Nr - 1);
    const int row2 = min(i0 + 32 + lane16, Nr - 1);
    const int row3 = min(i0 + 48 + lane16, Nr - 1);

    const _Float16* a0p = XH + (size_t)row0 * Kp + half * 8;
    const _Float16* a1p = XH + (size_t)row1 * Kp + half * 8;
    const _Float16* a2p = XH + (size_t)row2 * Kp + half * 8;
    const _Float16* a3p = XH + (size_t)row3 * Kp + half * 8;
    const _Float16* bp  = WT + (size_t)bcol * Kp + half * 8;

    v8f acc0 = {}, acc1 = {}, acc2 = {}, acc3 = {};

    for (int k0 = 0; k0 < Kp; k0 += 32) {
        const v8h b0 = *(const v8h*)(bp + k0);
        const v8h b1 = *(const v8h*)(bp + k0 + 16);
        const v16h bfrag = __builtin_shufflevector(b0, b1,
            0, 1, 2, 3, 4, 5, 6, 7, 8, 9, 10, 11, 12, 13, 14, 15);

        auto afrag = [&](const _Float16* ap) {
            const v8h lo = *(const v8h*)(ap + k0);
            const v8h hi = *(const v8h*)(ap + k0 + 16);
            return __builtin_shufflevector(lo, hi,
                0, 1, 2, 3, 4, 5, 6, 7, 8, 9, 10, 11, 12, 13, 14, 15);
        };

        acc0 = __builtin_amdgcn_wmma_f32_16x16x32_f16(false, afrag(a0p), false, bfrag,
                                                      (short)0, acc0, false, false);
        acc1 = __builtin_amdgcn_wmma_f32_16x16x32_f16(false, afrag(a1p), false, bfrag,
                                                      (short)0, acc1, false, false);
        acc2 = __builtin_amdgcn_wmma_f32_16x16x32_f16(false, afrag(a2p), false, bfrag,
                                                      (short)0, acc2, false, false);
        acc3 = __builtin_amdgcn_wmma_f32_16x16x32_f16(false, afrag(a3p), false, bfrag,
                                                      (short)0, acc3, false, false);
    }

    // Store D tiles.  lane16 = column, half selects M row block.
    const int col = j0 + lane16;
    if (col < F) {
        const float bv = bias ? bias[col] : 0.0f;
        auto store_tile = [&](const v8f& acc, int ibase) {
#pragma unroll
            for (int r = 0; r < 8; ++r) {
                const int row = ibase + r + half * 8;
                if (row < Nr) {
                    float v = acc[r] + bv;
                    if (relu) v = fmaxf(v, 0.0f);
                    Y[(size_t)row * F + col] = v;
                }
            }
        };
        store_tile(acc0, i0);
        store_tile(acc1, i0 + 16);
        store_tile(acc2, i0 + 32);
        store_tile(acc3, i0 + 48);
    }
}

// ------------------------- elementwise helpers ------------------------------
__global__ void fill_f32_kernel(float* __restrict__ p, float v, long long n)
{
    const long long i = blockIdx.x * (long long)blockDim.x + threadIdx.x;
    if (i < n) p[i] = v;
}

__global__ void fill_u32_kernel(unsigned* __restrict__ p, unsigned v, long long n)
{
    const long long i = blockIdx.x * (long long)blockDim.x + threadIdx.x;
    if (i < n) p[i] = v;
}

__global__ void bias_act_kernel(float* __restrict__ Y, const float* __restrict__ bias,
                                long long n, int F, int relu)
{
    const long long i = blockIdx.x * (long long)blockDim.x + threadIdx.x;
    if (i >= n) return;
    const int col = (int)(i % F);
    float v = Y[i] + bias[col];
    if (relu) v = fmaxf(v, 0.0f);
    Y[i] = v;
}

// ------------------------- GATv2 edge phase ---------------------------------
// Monotone float<->uint encoding so atomicMax(uint) implements float max.
__device__ __forceinline__ unsigned fenc(float f)
{
    const int i = __float_as_int(f);
    return (i >= 0) ? ((unsigned)i | 0x80000000u) : ~(unsigned)i;
}
__device__ __forceinline__ float fdec(unsigned u)
{
    const int i = (u & 0x80000000u) ? (int)(u & 0x7FFFFFFFu) : (int)~u;
    return __int_as_float(i);
}

// Self loops: edges [0,E) from edge_index, edges [E, E+Nn) are (i,i).
__device__ __forceinline__ void edge_src_dst(const int* ei, int E, int e, int& src, int& dst)
{
    if (e < E) { src = ei[e]; dst = ei[E + e]; }
    else       { src = dst = e - E; }
}

__device__ __forceinline__ float lrelu(float s)
{
    return (s > 0.0f) ? s : NEG_SLOPE * s;
}

// Pass 1: per (edge, head) logit = <leaky(xl[src]+xr[dst]), att[h]>; running max per dst.
// C is a multiple of 4 (C = 32); rows are 16B aligned -> b128 loads.
__global__ void edge_logit_max_kernel(const int* __restrict__ ei, int E, int Nn,
                                      const float* __restrict__ xl,
                                      const float* __restrict__ xr,
                                      const float* __restrict__ att, int H, int C,
                                      float* __restrict__ elog,
                                      unsigned* __restrict__ mbuf)
{
    const long long tid   = blockIdx.x * (long long)blockDim.x + threadIdx.x;
    const long long total = (long long)(E + Nn) * H;
    if (tid >= total) return;
    const int e = (int)(tid / H);
    const int h = (int)(tid % H);
    int src, dst; edge_src_dst(ei, E, e, src, dst);
    const int F = H * C;
    const float4* pl = (const float4*)(xl + (size_t)src * F + h * C);
    const float4* pr = (const float4*)(xr + (size_t)dst * F + h * C);
    const float4* pa = (const float4*)(att + h * C);
    float acc = 0.0f;
    for (int c = 0; c < (C >> 2); ++c) {
        const float4 a = pl[c], b = pr[c], w = pa[c];
        acc = fmaf(lrelu(a.x + b.x), w.x, acc);
        acc = fmaf(lrelu(a.y + b.y), w.y, acc);
        acc = fmaf(lrelu(a.z + b.z), w.z, acc);
        acc = fmaf(lrelu(a.w + b.w), w.w, acc);
    }
    elog[tid] = acc;
    atomicMax(&mbuf[(size_t)dst * H + h], fenc(acc));
}

// Pass 2: e = exp(logit - max[dst]); denominator accumulation.
__global__ void edge_exp_sum_kernel(const int* __restrict__ ei, int E, int Nn, int H,
                                    float* __restrict__ elog,
                                    const unsigned* __restrict__ mbuf,
                                    float* __restrict__ den)
{
    const long long tid   = blockIdx.x * (long long)blockDim.x + threadIdx.x;
    const long long total = (long long)(E + Nn) * H;
    if (tid >= total) return;
    const int e = (int)(tid / H);
    const int h = (int)(tid % H);
    int src, dst; edge_src_dst(ei, E, e, src, dst);
    (void)src;
    const float m  = fdec(mbuf[(size_t)dst * H + h]);
    const float ev = __expf(elog[tid] - m);
    elog[tid] = ev;
    atomicAdd(&den[(size_t)dst * H + h], ev);
}

// Pass 3: out[dst] += alpha * xl[src]  (per head slice).
__global__ void edge_scatter_kernel(const int* __restrict__ ei, int E, int Nn,
                                    int H, int C,
                                    const float* __restrict__ xl,
                                    const float* __restrict__ elog,
                                    const float* __restrict__ den,
                                    float* __restrict__ out)
{
    const long long tid   = blockIdx.x * (long long)blockDim.x + threadIdx.x;
    const long long total = (long long)(E + Nn) * H;
    if (tid >= total) return;
    const int e = (int)(tid / H);
    const int h = (int)(tid % H);
    int src, dst; edge_src_dst(ei, E, e, src, dst);
    const float alpha = elog[tid] / (den[(size_t)dst * H + h] + 1e-16f);
    const int F = H * C;
    const float4* pl = (const float4*)(xl + (size_t)src * F + h * C);
    __builtin_prefetch(pl, 0, 3);   // global_prefetch_b8 on the gathered row
    float* po = out + (size_t)dst * F + h * C;
    for (int c = 0; c < (C >> 2); ++c) {
        const float4 v = pl[c];
        atomicAdd(&po[4 * c + 0], alpha * v.x);
        atomicAdd(&po[4 * c + 1], alpha * v.y);
        atomicAdd(&po[4 * c + 2], alpha * v.z);
        atomicAdd(&po[4 * c + 3], alpha * v.w);
    }
}

// ------------------------- output head --------------------------------------
__global__ void head_kernel(const float* __restrict__ G,
                            const int* __restrict__ focal_idx,
                            const float* __restrict__ cl,
                            const float* __restrict__ clW,
                            const float* __restrict__ clb,
                            const float* __restrict__ fcW,
                            const float* __restrict__ fcb,
                            float* __restrict__ out,
                            int L, int HID, int OUT)
{
    __shared__ float comb[128];
    const int b = blockIdx.x;
    const int t = threadIdx.x;
    if (t < HID) {
        comb[t] = G[(size_t)focal_idx[b] * HID + t];
    } else if (t < 2 * HID) {
        const int j = t - HID;
        const float* p = cl + (size_t)b * L * 2;
        float mx = 0.0f, my = 0.0f;
        for (int l = 0; l < L; ++l) { mx += p[2 * l]; my += p[2 * l + 1]; }
        mx /= (float)L; my /= (float)L;
        comb[t] = mx * clW[j] + my * clW[HID + j] + clb[j];
    }
    __syncthreads();
    if (t < OUT) {
        float acc = fcb[t];
        const int D = 2 * HID;
        for (int k = 0; k < D; ++k) acc = fmaf(comb[k], fcW[k * OUT + t], acc);
        out[(size_t)b * OUT + t] = acc;
    }
}

// ------------------------- host orchestration -------------------------------
static inline void launch_cvt_rows(const float* src, _Float16* dst, int Nr, int K,
                                   hipStream_t s)
{
    const int Kp = padK(K);
    const long long n = (long long)Nr * Kp;
    cvt_rows_f16_kernel<<<(unsigned)((n + 255) / 256), 256, 0, s>>>(src, dst, n, K, Kp);
}

static inline void launch_cvt_wt(const float* W, _Float16* WT, int K, int F,
                                 hipStream_t s)
{
    const int Kp = padK(K);
    const long long n = (long long)F * Kp;
    cvt_wt_f16_kernel<<<(unsigned)((n + 255) / 256), 256, 0, s>>>(W, WT, n, K, F, Kp);
}

static inline void launch_linear(const _Float16* XH, const _Float16* WT, const float* b,
                                 float* Y, int Nr, int Kp, int F, int relu, hipStream_t s)
{
    dim3 grid((unsigned)((Nr + 63) / 64), (unsigned)((F + 63) / 64));
    linear_wmma_kernel<<<grid, 128, 0, s>>>(XH, WT, b, Y, Nr, Kp, F, relu);
}

static void run_gat(const float* Xin, int Nn, int Fin, int Fout, int H,
                    const float* Wl, const float* bl,
                    const float* Wr, const float* br,
                    const float* att, const float* bias,
                    const int* ei, int E,
                    _Float16* XH, _Float16* WT1, _Float16* WT2,
                    float* xl, float* xr, float* elog, unsigned* mb, float* den,
                    float* out, int relu, hipStream_t s)
{
    const int C  = Fout / H;
    const int Kp = padK(Fin);
    launch_cvt_rows(Xin, XH, Nn, Fin, s);
    launch_cvt_wt(Wl, WT1, Fin, Fout, s);
    launch_cvt_wt(Wr, WT2, Fin, Fout, s);
    launch_linear(XH, WT1, bl, xl, Nn, Kp, Fout, 0, s);
    launch_linear(XH, WT2, br, xr, Nn, Kp, Fout, 0, s);

    const long long nmh  = (long long)Nn * H;
    const long long nout = (long long)Nn * Fout;
    const int TB = 256;
    fill_u32_kernel<<<(unsigned)((nmh  + TB - 1) / TB), TB, 0, s>>>(mb,  0u,   nmh);
    fill_f32_kernel<<<(unsigned)((nmh  + TB - 1) / TB), TB, 0, s>>>(den, 0.0f, nmh);
    fill_f32_kernel<<<(unsigned)((nout + TB - 1) / TB), TB, 0, s>>>(out, 0.0f, nout);

    const long long teh = (long long)(E + Nn) * H;
    const unsigned  gb  = (unsigned)((teh + TB - 1) / TB);
    edge_logit_max_kernel<<<gb, TB, 0, s>>>(ei, E, Nn, xl, xr, att, H, C, elog, mb);
    edge_exp_sum_kernel  <<<gb, TB, 0, s>>>(ei, E, Nn, H, elog, mb, den);
    edge_scatter_kernel  <<<gb, TB, 0, s>>>(ei, E, Nn, H, C, xl, elog, den, out);
    bias_act_kernel<<<(unsigned)((nout + TB - 1) / TB), TB, 0, s>>>(out, bias, nout, Fout, relu);
}

extern "C" void kernel_launch(void* const* d_in, const int* in_sizes, int n_in,
                              void* d_out, int out_size, void* d_ws, size_t ws_size,
                              hipStream_t stream)
{
    // ----- inputs (setup_inputs order) -----
    const float* x           = (const float*)d_in[0];
    const int*   edge_index  = (const int*)  d_in[1];
    // d_in[2] lane_x, d_in[3] lane_edge_index, d_in[4] batch: dead code for the output
    const int*   focal_idx   = (const int*)  d_in[5];
    const float* centerlines = (const float*)d_in[6];
    const float* ae_W  = (const float*)d_in[7];
    const float* ae_b  = (const float*)d_in[8];
    const float* a1_Wl = (const float*)d_in[9];
    const float* a1_bl = (const float*)d_in[10];
    const float* a1_Wr = (const float*)d_in[11];
    const float* a1_br = (const float*)d_in[12];
    const float* a1_at = (const float*)d_in[13];
    const float* a1_b  = (const float*)d_in[14];
    const float* a2_Wl = (const float*)d_in[15];
    const float* a2_bl = (const float*)d_in[16];
    const float* a2_Wr = (const float*)d_in[17];
    const float* a2_br = (const float*)d_in[18];
    const float* a2_at = (const float*)d_in[19];
    const float* a2_b  = (const float*)d_in[20];
    const float* cl_W  = (const float*)d_in[35];
    const float* cl_b  = (const float*)d_in[36];
    const float* g_Wl  = (const float*)d_in[37];
    const float* g_bl  = (const float*)d_in[38];
    const float* g_Wr  = (const float*)d_in[39];
    const float* g_br  = (const float*)d_in[40];
    const float* g_at  = (const float*)d_in[41];
    const float* g_b   = (const float*)d_in[42];
    const float* fc_W  = (const float*)d_in[43];
    const float* fc_b  = (const float*)d_in[44];

    // ----- dimensions from in_sizes -----
    const int HID    = in_sizes[8];                 // 32
    const int H4     = in_sizes[14];                // 128
    const int HEADS  = H4 / HID;                    // 4
    const int SEQINC = in_sizes[7] / HID;           // 100
    const int N      = in_sizes[0] / SEQINC;        // 20000
    const int E      = in_sizes[1] / 2;             // 320000
    const int Bn     = in_sizes[5];                 // 64
    const int L      = in_sizes[6] / (Bn * 2);      // 50
    const int OUT    = in_sizes[44];                // 60

    const int KpMax = padK(SEQINC) > H4 ? padK(SEQINC) : H4;   // 128

    // ----- workspace arena (floats) -----
    float* ws = (float*)d_ws;
    size_t off = 0;
    auto alloc = [&](size_t nfloats) { float* p = ws + off; off += nfloats; return p; };
    float*    AX   = alloc((size_t)N * HID);              // agent embedding
    float*    XL   = alloc((size_t)N * H4);               // lin_l(x)
    float*    XR   = alloc((size_t)N * H4);               // lin_r(x)
    float*    H1   = alloc((size_t)N * H4);               // conv1 output
    float*    AF   = alloc((size_t)N * H4);               // agent_features
    float*    ELOG = alloc((size_t)(E + N) * HEADS);      // per-edge logits / exps
    float*    DEN  = alloc((size_t)N * HEADS);            // softmax denominators
    unsigned* MB   = (unsigned*)alloc((size_t)N * HEADS); // softmax running max (encoded)
    float*    G    = alloc((size_t)N * HID);              // global_features
    _Float16* XH   = (_Float16*)alloc(((size_t)N * KpMax + 1) / 2);  // f16 staged input
    _Float16* WT1  = (_Float16*)alloc(((size_t)H4 * KpMax + 1) / 2); // f16 transposed W
    _Float16* WT2  = (_Float16*)alloc(((size_t)H4 * KpMax + 1) / 2);
    (void)ws_size; (void)n_in; (void)out_size;

    // 1) agent embedding: relu(x.reshape(N,100) @ ae_W + ae_b)
    launch_cvt_rows(x, XH, N, SEQINC, stream);
    launch_cvt_wt(ae_W, WT1, SEQINC, HID, stream);
    launch_linear(XH, WT1, ae_b, AX, N, padK(SEQINC), HID, 1, stream);

    // 2) agent GATv2 conv1 + relu
    run_gat(AX, N, HID, H4, HEADS, a1_Wl, a1_bl, a1_Wr, a1_br, a1_at, a1_b,
            edge_index, E, XH, WT1, WT2, XL, XR, ELOG, MB, DEN, H1, 1, stream);

    // 3) agent GATv2 conv2 + relu -> agent_features
    run_gat(H1, N, H4, H4, HEADS, a2_Wl, a2_bl, a2_Wr, a2_br, a2_at, a2_b,
            edge_index, E, XH, WT1, WT2, XL, XR, ELOG, MB, DEN, AF, 1, stream);

    // (map branch skipped: its result is dead code in the reference output)

    // 4) global graph (heads=1) + relu -> global_features
    run_gat(AF, N, H4, HID, 1, g_Wl, g_bl, g_Wr, g_br, g_at, g_b,
            edge_index, E, XH, WT1, WT2, XL, XR, ELOG, MB, DEN, G, 1, stream);

    // 5) head: gather focal, centerline mean-linear, concat, fc
    head_kernel<<<(unsigned)Bn, 64, 0, stream>>>(G, focal_idx, centerlines,
                                                 cl_W, cl_b, fc_W, fc_b,
                                                 (float*)d_out, L, HID, OUT);
}
